// BitLinear158_24438363914262
// MI455X (gfx1250) — matrix operand order repaired and checked by reading, not compiled, not run on hardware
//
#include <hip/hip_runtime.h>

// ---------------------------------------------------------------------------
// BitLinear158 forward, MI455X (gfx1250, wave32, WMMA).
//   out[M,N] = x[M,K] * ternary(W)[N,K]^T, gamma = mean|W| + eps.
// M=8192, N=16384, K=4096: 1.1 TFLOP vs ~0.92 GB -> compute-bound on the
// WMMA pipe (v_wmma_f32_16x16x32_bf16; ternary weights exact in bf16).
// Fast path: preconvert x->bf16 and W->ternary bf16 into d_ws, then a
// double-buffered 128x256 tile GEMM fed by async global->LDS copies
// (ASYNCcnt), 64x64 per wave (16 accumulators) for a 1:1 ds:wmma ratio.
// Steady-state loop is branch-free (last stage peeled).
// Fallback (small ws): fused-quantize GEMM.
// ---------------------------------------------------------------------------

typedef __attribute__((ext_vector_type(16))) __bf16 v16bf;
typedef __attribute__((ext_vector_type(8)))  __bf16 v8bf;
typedef __attribute__((ext_vector_type(4)))  __bf16 v4bf;
typedef __attribute__((ext_vector_type(8)))  float  v8f;
typedef __attribute__((ext_vector_type(4)))  float  v4f;

// Pointer types expected by the async global->LDS builtin (per hipcc
// diagnostic: parameter is 'int __attribute__((vector_size(16))) AS*').
typedef int v4i_vs __attribute__((vector_size(16)));
#define AS1_V4I(p) ((__attribute__((address_space(1))) v4i_vs*)(p))
#define AS3_V4I(p) ((__attribute__((address_space(3))) v4i_vs*)(p))

#if __has_builtin(__builtin_amdgcn_global_load_async_to_lds_b128)
#define USE_ASYNC_LDS 1
#else
#define USE_ASYNC_LDS 0
#endif

#if __has_builtin(__builtin_amdgcn_s_wait_asynccnt)
#define WAIT_ASYNC(n) __builtin_amdgcn_s_wait_asynccnt(n)
#else
#define WAIT_ASYNC(n) asm volatile("s_wait_asynccnt %0" ::"i"(n))
#endif

// ---------------- gamma = mean(|W|) + eps : deterministic 2-pass reduction --

__global__ __launch_bounds__(256)
void absum_partial_kernel(const float* __restrict__ w, float* __restrict__ part,
                          size_t n4) {
  __shared__ float red[256];
  const size_t stride = (size_t)gridDim.x * 256u;
  float s = 0.0f;
  for (size_t i = (size_t)blockIdx.x * 256u + threadIdx.x; i < n4; i += stride) {
    v4f v = *(const v4f*)(w + i * 4);
    s += fabsf(v[0]) + fabsf(v[1]) + fabsf(v[2]) + fabsf(v[3]);
  }
  red[threadIdx.x] = s;
  __syncthreads();
  for (int off = 128; off > 0; off >>= 1) {
    if ((int)threadIdx.x < off) red[threadIdx.x] += red[threadIdx.x + off];
    __syncthreads();
  }
  if (threadIdx.x == 0) part[blockIdx.x] = red[0];
}

__global__ __launch_bounds__(256)
void absum_final_kernel(const float* __restrict__ part, float* __restrict__ gamma,
                        float inv_n) {
  __shared__ float red[256];
  const int t = threadIdx.x;
  red[t] = part[t] + part[t + 256] + part[t + 512] + part[t + 768];
  __syncthreads();
  for (int off = 128; off > 0; off >>= 1) {
    if (t < off) red[t] += red[t + off];
    __syncthreads();
  }
  if (t == 0) gamma[0] = red[0] * inv_n + 1e-6f;
}

// ---------------- precision conversion pre-passes ---------------------------

__global__ __launch_bounds__(256)
void cvt_x_kernel(const float* __restrict__ x, __bf16* __restrict__ xb, size_t n4) {
  const size_t stride = (size_t)gridDim.x * 256u;
  for (size_t i = (size_t)blockIdx.x * 256u + threadIdx.x; i < n4; i += stride) {
    v4f v = *(const v4f*)(x + i * 4);
    v4bf b;
    b[0] = (__bf16)v[0]; b[1] = (__bf16)v[1];
    b[2] = (__bf16)v[2]; b[3] = (__bf16)v[3];
    *(v4bf*)(xb + i * 4) = b;
  }
}

__global__ __launch_bounds__(256)
void quant_w_kernel(const float* __restrict__ w, __bf16* __restrict__ wb,
                    const float* __restrict__ gammap, size_t n4) {
  const float thr = 0.5f * gammap[0];
  const size_t stride = (size_t)gridDim.x * 256u;
  for (size_t i = (size_t)blockIdx.x * 256u + threadIdx.x; i < n4; i += stride) {
    v4f v = *(const v4f*)(w + i * 4);
    v4bf b;
#pragma unroll
    for (int e = 0; e < 4; ++e) {
      float q = (fabsf(v[e]) >= thr) ? copysignf(1.0f, v[e]) : 0.0f;
      b[e] = (__bf16)q;  // exact
    }
    *(v4bf*)(wb + i * 4) = b;
  }
}

// ---------------- shared WMMA tile core (64x64 per wave) --------------------

#define TM2  128
#define TN2  256
#define TK2  32
#define LDT2 40   // 80 B row stride = 5*16 B -> every fragment chunk 16B-aligned

__device__ __forceinline__ void wmma_compute_64x64(const __bf16* __restrict__ A,
                                                   const __bf16* __restrict__ B,
                                                   int wm, int wn, int l16, int half,
                                                   v8f acc[4][4]) {
  v16bf a[4], b[4];
#pragma unroll
  for (int i = 0; i < 4; ++i) {
    // A 16x32 frag: halves[0..7]=K[half*8..], halves[8..15]=K[16+half*8..]
    const __bf16* base = A + (wm + 16 * i + l16) * LDT2 + half * 8;
    v8bf lo = *(const v8bf*)base;
    v8bf hh = *(const v8bf*)(base + 16);
    a[i] = __builtin_shufflevector(lo, hh, 0,1,2,3,4,5,6,7,8,9,10,11,12,13,14,15);
  }
#pragma unroll
  for (int j = 0; j < 4; ++j) {
    // B 32x16 frag: lane = column, halves[0..7]=K[half*8..], halves[8..15]=K[16+half*8..]
    const __bf16* base = B + (wn + 16 * j + l16) * LDT2 + half * 8;
    v8bf lo = *(const v8bf*)base;
    v8bf hh = *(const v8bf*)(base + 16);
    b[j] = __builtin_shufflevector(lo, hh, 0,1,2,3,4,5,6,7,8,9,10,11,12,13,14,15);
  }
#pragma unroll
  for (int i = 0; i < 4; ++i)
#pragma unroll
    for (int j = 0; j < 4; ++j)
      acc[i][j] = __builtin_amdgcn_wmma_f32_16x16x32_bf16(
          false, a[i], false, b[j], (short)0, acc[i][j], false, false);
}

// ---------------- fast-path GEMM: preconverted bf16 operands ----------------
// 256 threads = 8 wave32 in 2x4 grid, workgroup tile 128(M) x 256(N), TK=32.

__global__ __launch_bounds__(256)
void bitlinear_wmma_pre_kernel(const __bf16* __restrict__ xb,
                               const __bf16* __restrict__ wb,
                               float* __restrict__ out,
                               int M, int N, int K) {
#if USE_ASYNC_LDS
  __shared__ __bf16 Als[2][TM2 * LDT2];   // 2 x 10 KB
  __shared__ __bf16 Bls[2][TN2 * LDT2];   // 2 x 20 KB
#else
  __shared__ __bf16 Als[1][TM2 * LDT2];
  __shared__ __bf16 Bls[1][TN2 * LDT2];
#endif
  const int tid  = threadIdx.x;
  const int m0   = blockIdx.y * TM2;
  const int n0   = blockIdx.x * TN2;
  const int wid  = tid >> 5;
  const int lane = tid & 31;
  const int l16  = lane & 15;
  const int half = lane >> 4;
  const int wm   = (wid >> 2) * 64;
  const int wn   = (wid & 3) * 64;

  // staging: one 16B chunk (8 bf16) per (row, c); A: 512 chunks, B: 1024 chunks
  const int srow = tid >> 2;
  const int sc   = (tid & 3) << 3;

  v8f acc[4][4] = {};
  const int nstage = K / TK2;

#if USE_ASYNC_LDS
  // --- CDNA5 async global->LDS copies, double-buffered (ASYNCcnt) ---
  auto issue = [&](int k0, int buf) {
#pragma unroll
    for (int t = 0; t < 2; ++t) {
      const int row = srow + t * 64;
      __builtin_amdgcn_global_load_async_to_lds_b128(
          AS1_V4I(xb + (size_t)(m0 + row) * K + k0 + sc),
          AS3_V4I(&Als[buf][row * LDT2 + sc]),
          0, 0);
    }
#pragma unroll
    for (int t = 0; t < 4; ++t) {
      const int row = srow + t * 64;
      __builtin_amdgcn_global_load_async_to_lds_b128(
          AS1_V4I(wb + (size_t)(n0 + row) * K + k0 + sc),
          AS3_V4I(&Bls[buf][row * LDT2 + sc]),
          0, 0);
    }
  };
  issue(0, 0);
  // branch-free steady state; last stage peeled
  for (int s = 0; s < nstage - 1; ++s) {
    issue((s + 1) * TK2, (s + 1) & 1);
    WAIT_ASYNC(6);          // 6 newly issued outstanding; stage s complete
    __syncthreads();
    wmma_compute_64x64(&Als[s & 1][0], &Bls[s & 1][0], wm, wn, l16, half, acc);
    __syncthreads();        // buffer free before it is overwritten
  }
  WAIT_ASYNC(0);
  __syncthreads();
  wmma_compute_64x64(&Als[(nstage - 1) & 1][0], &Bls[(nstage - 1) & 1][0],
                     wm, wn, l16, half, acc);
#else
  // --- fallback: pipelined global->VGPR->LDS, single buffer ---
  uint4 ra[2], rb[4];
  auto prefetch = [&](int k0) {
#pragma unroll
    for (int t = 0; t < 2; ++t)
      ra[t] = *(const uint4*)(xb + (size_t)(m0 + srow + t * 64) * K + k0 + sc);
#pragma unroll
    for (int t = 0; t < 4; ++t)
      rb[t] = *(const uint4*)(wb + (size_t)(n0 + srow + t * 64) * K + k0 + sc);
  };
  prefetch(0);
  for (int s = 0; s < nstage; ++s) {
#pragma unroll
    for (int t = 0; t < 2; ++t)
      *(uint4*)&Als[0][(srow + t * 64) * LDT2 + sc] = ra[t];
#pragma unroll
    for (int t = 0; t < 4; ++t)
      *(uint4*)&Bls[0][(srow + t * 64) * LDT2 + sc] = rb[t];
    __syncthreads();
    if (s + 1 < nstage) prefetch((s + 1) * TK2);  // overlaps with compute
    wmma_compute_64x64(&Als[0][0], &Bls[0][0], wm, wn, l16, half, acc);
    __syncthreads();
  }
#endif

  // D layout: VGPR r holds (M = r + 8*half, N = l16)
#pragma unroll
  for (int i = 0; i < 4; ++i)
#pragma unroll
    for (int j = 0; j < 4; ++j) {
      const int mbase = m0 + wm + 16 * i + 8 * half;
      const int n     = n0 + wn + 16 * j + l16;
#pragma unroll
      for (int r = 0; r < 8; ++r)
        out[(size_t)(mbase + r) * N + n] = acc[i][j][r];
    }
}

// ---------------- fallback fused-quantize GEMM (small ws) -------------------
// Workgroup tile 128x128, 8 waves of 64x32, K staged 64 at a time.

#define TM  128
#define TN  128
#define TK  64
#define LDT 72

__global__ __launch_bounds__(256)
void bitlinear_wmma_fused_kernel(const float* __restrict__ x,
                                 const float* __restrict__ w,
                                 const float* __restrict__ gammap,
                                 float* __restrict__ out,
                                 int M, int N, int K) {
  __shared__ __bf16 Als[TM * LDT];
  __shared__ __bf16 Bls[TN * LDT];

  const float thr = 0.5f * gammap[0];
  const int tid  = threadIdx.x;
  const int m0   = blockIdx.y * TM;
  const int n0   = blockIdx.x * TN;
  const int wid  = tid >> 5;
  const int lane = tid & 31;
  const int l16  = lane & 15;
  const int half = lane >> 4;
  const int wm   = (wid >> 2) * 64;
  const int wn   = (wid & 3) * 32;

  v8f acc[4][2] = {};

  const int srow = tid >> 4;
  const int skk  = (tid & 15) << 2;

  for (int k0 = 0; k0 < K; k0 += TK) {
#pragma unroll
    for (int t = 0; t < 8; ++t) {
      const int row = srow + t * 16;
      v4f v = *(const v4f*)(x + (size_t)(m0 + row) * K + (k0 + skk));
      v4bf b;
      b[0] = (__bf16)v[0]; b[1] = (__bf16)v[1];
      b[2] = (__bf16)v[2]; b[3] = (__bf16)v[3];
      *(v4bf*)&Als[row * LDT + skk] = b;
    }
#pragma unroll
    for (int t = 0; t < 8; ++t) {
      const int row = srow + t * 16;
      v4f v = *(const v4f*)(w + (size_t)(n0 + row) * K + (k0 + skk));
      v4bf b;
#pragma unroll
      for (int e = 0; e < 4; ++e) {
        float q = (fabsf(v[e]) >= thr) ? copysignf(1.0f, v[e]) : 0.0f;
        b[e] = (__bf16)q;
      }
      *(v4bf*)&Bls[row * LDT + skk] = b;
    }
    __syncthreads();

#pragma unroll
    for (int ks = 0; ks < TK; ks += 32) {
      v16bf afrag[4];
#pragma unroll
      for (int i = 0; i < 4; ++i) {
        const __bf16* base = &Als[(wm + 16 * i + l16) * LDT + ks + half * 8];
        v8bf lo = *(const v8bf*)base;
        v8bf hh = *(const v8bf*)(base + 16);
        afrag[i] = __builtin_shufflevector(lo, hh,
            0,1,2,3,4,5,6,7,8,9,10,11,12,13,14,15);
      }
      v16bf bfrag[2];
#pragma unroll
      for (int j = 0; j < 2; ++j) {
        const __bf16* base = &Bls[(wn + 16 * j + l16) * LDT + ks + half * 8];
        v8bf lo = *(const v8bf*)base;
        v8bf hh = *(const v8bf*)(base + 16);
        bfrag[j] = __builtin_shufflevector(lo, hh,
            0,1,2,3,4,5,6,7,8,9,10,11,12,13,14,15);
      }
#pragma unroll
      for (int i = 0; i < 4; ++i)
#pragma unroll
        for (int j = 0; j < 2; ++j)
          acc[i][j] = __builtin_amdgcn_wmma_f32_16x16x32_bf16(
              false, afrag[i], false, bfrag[j],
              (short)0, acc[i][j], false, false);
    }
    __syncthreads();
  }

#pragma unroll
  for (int i = 0; i < 4; ++i)
#pragma unroll
    for (int j = 0; j < 2; ++j) {
      const int mbase = m0 + wm + 16 * i + 8 * half;
      const int n     = n0 + wn + 16 * j + l16;
#pragma unroll
      for (int r = 0; r < 8; ++r)
        out[(size_t)(mbase + r) * N + n] = acc[i][j][r];
    }
}

// ---------------------------------------------------------------------------

extern "C" void kernel_launch(void* const* d_in, const int* in_sizes, int n_in,
                              void* d_out, int out_size, void* d_ws, size_t ws_size,
                              hipStream_t stream) {
  const float* x = (const float*)d_in[0];      // [B,S,K] f32
  const float* w = (const float*)d_in[1];      // [N,K]   f32
  float* out = (float*)d_out;                  // [B,S,N] f32

  const int K = 4096;
  const int M = in_sizes[0] / K;               // 8192
  const int N = in_sizes[1] / K;               // 16384

  float* part  = (float*)d_ws;                 // 1024 partials
  float* gamma = part + 1024;                  // 1 scalar @ +4096 B

  const size_t nw4 = (size_t)N * (size_t)K / 4;
  absum_partial_kernel<<<1024, 256, 0, stream>>>(w, part, nw4);
  absum_final_kernel<<<1, 256, 0, stream>>>(part, gamma,
                                            1.0f / ((float)N * (float)K));

  const size_t need = 8192 + (size_t)M * K * 2 + (size_t)N * K * 2;
  if (ws_size >= need && (M % TM2) == 0 && (N % TN2) == 0) {
    __bf16* xb = (__bf16*)((char*)d_ws + 8192);
    __bf16* wb = xb + (size_t)M * K;
    cvt_x_kernel<<<4096, 256, 0, stream>>>(x, xb, (size_t)M * K / 4);
    quant_w_kernel<<<8192, 256, 0, stream>>>(w, wb, gamma, (size_t)N * K / 4);
    dim3 grid(N / TN2, M / TM2);
    bitlinear_wmma_pre_kernel<<<grid, 256, 0, stream>>>(xb, wb, out, M, N, K);
  } else {
    dim3 grid(N / TN, M / TM);
    bitlinear_wmma_fused_kernel<<<grid, 256, 0, stream>>>(x, w, gamma, out,
                                                          M, N, K);
  }
}
